// CustomLayer_35682588295215
// MI455X (gfx1250) — compile-verified, matrix operand order checked by reading, not run on hardware
//
#include <hip/hip_runtime.h>

// ---------------------------------------------------------------------------
// out[m, o] = sum_k X[m,k] * W[o,k];  X: 8192x4096 f32, W: COO->dense 4096x4096.
// Pipeline: zero W -> atomic scatter -> one-pass f32->bf16(hi/lo) split of X and W
//           -> bf16x3 split-precision WMMA GEMM with pure-copy LDS staging
//           (async global->LDS via gfx1250 builtin when available).
// Fallback (small ws): convert-in-loop GEMM needing only 64 MB scratch.
// ---------------------------------------------------------------------------

typedef __attribute__((ext_vector_type(16))) __bf16 v16bf;
typedef __attribute__((ext_vector_type(8)))  __bf16 v8bf;
typedef __attribute__((ext_vector_type(8)))  float  v8f;
typedef __attribute__((ext_vector_type(4)))  float  v4f;

#define N_OUT   4096
#define K_DIM   4096
#define M_TOTAL 8192
#define BM      128
#define BN      128
#define BK      32
#define LDST    40        // LDS row stride in halves (80 B: 16B-aligned, non-pow2)

#if defined(__AMDGCN__) && __has_builtin(__builtin_amdgcn_global_load_async_to_lds_b128) && __has_builtin(__builtin_amdgcn_s_wait_asynccnt)
#define USE_ASYNC_LDS 1
#endif

// ------------------------- helpers -----------------------------------------

__device__ __forceinline__ void split_bits(float x, unsigned int& hb, unsigned int& lb) {
  unsigned int u = __builtin_bit_cast(unsigned int, x);
  hb = (u + 0x7FFFu + ((u >> 16) & 1u)) >> 16;          // rne bf16(x)
  float hf = __builtin_bit_cast(float, hb << 16);
  float rest = x - hf;
  unsigned int v = __builtin_bit_cast(unsigned int, rest);
  lb = (v + 0x7FFFu + ((v >> 16) & 1u)) >> 16;          // rne bf16(x - hi)
}

__device__ __forceinline__ void split_bf16(float x, __bf16& hi, __bf16& lo) {
  unsigned int hb, lb;
  split_bits(x, hb, lb);
  hi = __builtin_bit_cast(__bf16, (unsigned short)hb);
  lo = __builtin_bit_cast(__bf16, (unsigned short)lb);
}

// ISA 16-bit A/B fragment: lane L holds row/col (L%16); h = L/16 selects the
// K half-chunks {8h..8h+7} (elems 0..7) and {16+8h..23+8h} (elems 8..15).
__device__ __forceinline__ v16bf load_frag(const __bf16* rowptr, int h) {
  v8bf a = *(const v8bf*)(rowptr + 8 * h);
  v8bf b = *(const v8bf*)(rowptr + 16 + 8 * h);
  return __builtin_shufflevector(a, b, 0, 1, 2, 3, 4, 5, 6, 7,
                                       8, 9, 10, 11, 12, 13, 14, 15);
}

// stage one 16-byte chunk global->LDS (async path when available)
__device__ __forceinline__ void stage16(const __bf16* gsrc, __bf16* ldst) {
#if defined(USE_ASYNC_LDS)
  typedef int v4i_vs __attribute__((vector_size(16)));
  typedef __attribute__((address_space(1))) v4i_vs* gp_t;   // global (AS1)
  typedef __attribute__((address_space(3))) v4i_vs* lp_t;   // LDS (AS3)
  __builtin_amdgcn_global_load_async_to_lds_b128((gp_t)gsrc, (lp_t)ldst, 0, 0);
#else
  *(uint4*)ldst = *(const uint4*)gsrc;
#endif
}

__device__ __forceinline__ void stage_fence() {
#if defined(USE_ASYNC_LDS)
  __builtin_amdgcn_s_wait_asynccnt(0);
#endif
  __syncthreads();
}

// ------------------------- kernel 1: zero W --------------------------------

__global__ void zero_w_kernel(float4* __restrict__ w4, int n4) {
  int i = blockIdx.x * blockDim.x + threadIdx.x;
  if (i < n4) w4[i] = make_float4(0.f, 0.f, 0.f, 0.f);
}

// ------------------------- kernel 2: COO scatter ---------------------------

__global__ void scatter_kernel(const float* __restrict__ vals,
                               const int* __restrict__ cols,
                               const int* __restrict__ rows,
                               float* __restrict__ W, int nnz) {
  int i = blockIdx.x * blockDim.x + threadIdx.x;
  if (i < nnz)
    atomicAdd(&W[(size_t)rows[i] * K_DIM + cols[i]], vals[i]);
}

// ---------------- kernel 3: one-pass f32 -> bf16 hi/lo split ---------------

__global__ void convert_split_kernel(const float* __restrict__ src,
                                     __bf16* __restrict__ hi,
                                     __bf16* __restrict__ lo, int n4) {
  int i = blockIdx.x * blockDim.x + threadIdx.x;
  if (i >= n4) return;
  v4f v = ((const v4f*)src)[i];
  unsigned int hb[4], lb[4];
  #pragma unroll
  for (int j = 0; j < 4; ++j) split_bits(v[j], hb[j], lb[j]);
  uint2 hp, lp;
  hp.x = hb[0] | (hb[1] << 16);  hp.y = hb[2] | (hb[3] << 16);
  lp.x = lb[0] | (lb[1] << 16);  lp.y = lb[2] | (lb[3] << 16);
  ((uint2*)hi)[i] = hp;
  ((uint2*)lo)[i] = lp;
}

// -------------- kernel 4 (fast path): pure-copy WMMA GEMM ------------------
// C[m,n] = sum_k X[m,k]*W[n,k]; acc += AhBh + AhBl + AlBh (f32 accumulators).

__global__ __launch_bounds__(256)
void gemm_bf16_kernel(const __bf16* __restrict__ Xhi, const __bf16* __restrict__ Xlo,
                      const __bf16* __restrict__ Whi, const __bf16* __restrict__ Wlo,
                      float* __restrict__ out) {
  __shared__ __bf16 sXhi[BM][LDST];
  __shared__ __bf16 sXlo[BM][LDST];
  __shared__ __bf16 sWhi[BN][LDST];
  __shared__ __bf16 sWlo[BN][LDST];

  const int tid   = threadIdx.x;
  const int lane  = tid & 31;
  const int wave  = tid >> 5;
  const int r     = lane & 15;
  const int h     = lane >> 4;
  const int waveM = wave & 3;     // 4 x 32-row strips
  const int waveN = wave >> 2;    // 2 x 64-col strips

  const int mBase = blockIdx.x * BM;
  const int nBase = blockIdx.y * BN;

  // staging map: 128 rows x 4 chunks(16B) = 512 chunks per plane; 2 per thread
  const int row0 = (tid + 0)   >> 2, c0 = (tid + 0)   & 3;
  const int row1 = (tid + 256) >> 2, c1 = (tid + 256) & 3;

  v8f acc[2][4] = {};

  #pragma unroll 1
  for (int kt = 0; kt < K_DIM / BK; ++kt) {
    const int k0 = kt * BK;

    stage16(Xhi + (size_t)(mBase + row0) * K_DIM + k0 + c0 * 8, &sXhi[row0][c0 * 8]);
    stage16(Xhi + (size_t)(mBase + row1) * K_DIM + k0 + c1 * 8, &sXhi[row1][c1 * 8]);
    stage16(Xlo + (size_t)(mBase + row0) * K_DIM + k0 + c0 * 8, &sXlo[row0][c0 * 8]);
    stage16(Xlo + (size_t)(mBase + row1) * K_DIM + k0 + c1 * 8, &sXlo[row1][c1 * 8]);
    stage16(Whi + (size_t)(nBase + row0) * K_DIM + k0 + c0 * 8, &sWhi[row0][c0 * 8]);
    stage16(Whi + (size_t)(nBase + row1) * K_DIM + k0 + c1 * 8, &sWhi[row1][c1 * 8]);
    stage16(Wlo + (size_t)(nBase + row0) * K_DIM + k0 + c0 * 8, &sWlo[row0][c0 * 8]);
    stage16(Wlo + (size_t)(nBase + row1) * K_DIM + k0 + c1 * 8, &sWlo[row1][c1 * 8]);
    stage_fence();

    v16bf aHi[2], aLo[2];
    #pragma unroll
    for (int mi = 0; mi < 2; ++mi) {
      const int rowA = waveM * 32 + mi * 16 + r;
      aHi[mi] = load_frag(&sXhi[rowA][0], h);
      aLo[mi] = load_frag(&sXlo[rowA][0], h);
    }
    #pragma unroll
    for (int ni = 0; ni < 4; ++ni) {
      const int rowB = waveN * 64 + ni * 16 + r;
      v16bf bHi = load_frag(&sWhi[rowB][0], h);
      v16bf bLo = load_frag(&sWlo[rowB][0], h);
      #pragma unroll
      for (int mi = 0; mi < 2; ++mi) {
        acc[mi][ni] = __builtin_amdgcn_wmma_f32_16x16x32_bf16(
            false, aHi[mi], false, bHi, (short)0, acc[mi][ni], false, false);
        acc[mi][ni] = __builtin_amdgcn_wmma_f32_16x16x32_bf16(
            false, aHi[mi], false, bLo, (short)0, acc[mi][ni], false, false);
        acc[mi][ni] = __builtin_amdgcn_wmma_f32_16x16x32_bf16(
            false, aLo[mi], false, bHi, (short)0, acc[mi][ni], false, false);
      }
    }
    __syncthreads();
  }

  #pragma unroll
  for (int mi = 0; mi < 2; ++mi)
    #pragma unroll
    for (int ni = 0; ni < 4; ++ni) {
      const int col = nBase + waveN * 64 + ni * 16 + r;
      #pragma unroll
      for (int v = 0; v < 8; ++v) {
        const int row = mBase + waveM * 32 + mi * 16 + 8 * h + v;
        out[(size_t)row * N_OUT + col] = acc[mi][ni][v];
      }
    }
}

// -------- kernel 5 (fallback, small ws): convert-in-loop WMMA GEMM ---------

__global__ __launch_bounds__(256)
void spmm_wmma_fallback_kernel(const float* __restrict__ X,
                               const float* __restrict__ W,
                               float* __restrict__ out) {
  __shared__ __bf16 sXhi[BM][LDST];
  __shared__ __bf16 sXlo[BM][LDST];
  __shared__ __bf16 sWhi[BN][LDST];
  __shared__ __bf16 sWlo[BN][LDST];

  const int tid   = threadIdx.x;
  const int lane  = tid & 31;
  const int wave  = tid >> 5;
  const int r     = lane & 15;
  const int h     = lane >> 4;
  const int waveM = wave & 3;
  const int waveN = wave >> 2;

  const int mBase = blockIdx.x * BM;
  const int nBase = blockIdx.y * BN;

  v8f acc[2][4] = {};

  #pragma unroll 1
  for (int kt = 0; kt < K_DIM / BK; ++kt) {
    const int k0 = kt * BK;
    #pragma unroll
    for (int it = 0; it < 4; ++it) {
      int idx = tid + it * 256, row = idx >> 3, q = idx & 7;
      v4f v = *(const v4f*)(X + (size_t)(mBase + row) * K_DIM + k0 + q * 4);
      #pragma unroll
      for (int j = 0; j < 4; ++j)
        split_bf16(v[j], sXhi[row][q * 4 + j], sXlo[row][q * 4 + j]);
    }
    #pragma unroll
    for (int it = 0; it < 4; ++it) {
      int idx = tid + it * 256, row = idx >> 3, q = idx & 7;
      v4f v = *(const v4f*)(W + (size_t)(nBase + row) * K_DIM + k0 + q * 4);
      #pragma unroll
      for (int j = 0; j < 4; ++j)
        split_bf16(v[j], sWhi[row][q * 4 + j], sWlo[row][q * 4 + j]);
    }
    __syncthreads();

    v16bf aHi[2], aLo[2];
    #pragma unroll
    for (int mi = 0; mi < 2; ++mi) {
      const int rowA = waveM * 32 + mi * 16 + r;
      aHi[mi] = load_frag(&sXhi[rowA][0], h);
      aLo[mi] = load_frag(&sXlo[rowA][0], h);
    }
    #pragma unroll
    for (int ni = 0; ni < 4; ++ni) {
      const int rowB = waveN * 64 + ni * 16 + r;
      v16bf bHi = load_frag(&sWhi[rowB][0], h);
      v16bf bLo = load_frag(&sWlo[rowB][0], h);
      #pragma unroll
      for (int mi = 0; mi < 2; ++mi) {
        acc[mi][ni] = __builtin_amdgcn_wmma_f32_16x16x32_bf16(
            false, aHi[mi], false, bHi, (short)0, acc[mi][ni], false, false);
        acc[mi][ni] = __builtin_amdgcn_wmma_f32_16x16x32_bf16(
            false, aHi[mi], false, bLo, (short)0, acc[mi][ni], false, false);
        acc[mi][ni] = __builtin_amdgcn_wmma_f32_16x16x32_bf16(
            false, aLo[mi], false, bHi, (short)0, acc[mi][ni], false, false);
      }
    }
    __syncthreads();
  }

  #pragma unroll
  for (int mi = 0; mi < 2; ++mi)
    #pragma unroll
    for (int ni = 0; ni < 4; ++ni) {
      const int col = nBase + waveN * 64 + ni * 16 + r;
      #pragma unroll
      for (int v = 0; v < 8; ++v) {
        const int row = mBase + waveM * 32 + mi * 16 + 8 * h + v;
        out[(size_t)row * N_OUT + col] = acc[mi][ni][v];
      }
    }
}

// ------------------------- launch ------------------------------------------

extern "C" void kernel_launch(void* const* d_in, const int* in_sizes, int n_in,
                              void* d_out, int out_size, void* d_ws, size_t ws_size,
                              hipStream_t stream) {
  const float* x       = (const float*)d_in[0];
  const float* values  = (const float*)d_in[1];
  const int*   col_idx = (const int*)d_in[2];
  const int*   row_ids = (const int*)d_in[3];
  float*       out     = (float*)d_out;

  const int nnz = in_sizes[1];

  // workspace layout
  const size_t szWf32 = (size_t)N_OUT * K_DIM * sizeof(float);     //  64 MB
  const size_t szWbf  = (size_t)N_OUT * K_DIM * 2;                 //  32 MB
  const size_t szXbf  = (size_t)M_TOTAL * K_DIM * 2;               //  64 MB
  const size_t need   = szWf32 + 2 * szWbf + 2 * szXbf;            // 256 MB

  char*  ws  = (char*)d_ws;
  float* W   = (float*)ws;

  // 1) zero dense W, 2) scatter COO (duplicates accumulate)
  const int n4w = (N_OUT * K_DIM) / 4;
  zero_w_kernel<<<(n4w + 255) / 256, 256, 0, stream>>>((float4*)W, n4w);
  scatter_kernel<<<(nnz + 255) / 256, 256, 0, stream>>>(values, col_idx, row_ids, W, nnz);

  dim3 grid(M_TOTAL / BM, N_OUT / BN);   // (64, 32)

  if (ws_size >= need) {
    __bf16* Whi = (__bf16*)(ws + szWf32);
    __bf16* Wlo = (__bf16*)(ws + szWf32 + szWbf);
    __bf16* Xhi = (__bf16*)(ws + szWf32 + 2 * szWbf);
    __bf16* Xlo = (__bf16*)(ws + szWf32 + 2 * szWbf + szXbf);

    // 3) one-pass hi/lo splits
    const int n4x = (M_TOTAL * K_DIM) / 4;
    convert_split_kernel<<<(n4w + 255) / 256, 256, 0, stream>>>(W, Whi, Wlo, n4w);
    convert_split_kernel<<<(n4x + 255) / 256, 256, 0, stream>>>(x, Xhi, Xlo, n4x);

    // 4) pure-copy WMMA GEMM
    gemm_bf16_kernel<<<grid, 256, 0, stream>>>(Xhi, Xlo, Whi, Wlo, out);
  } else {
    // fallback: convert inside the GEMM loop (only needs dense W scratch)
    spmm_wmma_fallback_kernel<<<grid, 256, 0, stream>>>(x, W, out);
  }
}
